// Siamese_80487687127057
// MI455X (gfx1250) — matrix operand order repaired and checked by reading, not compiled, not run on hardware
//
#include <hip/hip_runtime.h>
#include <hip/hip_bf16.h>

// ---------------------------------------------------------------------------
// Siamese / prototypical head for MI455X (gfx1250, wave32, WMMA bf16).
//
// Roofline: GEMM = 26.8 GFLOP; min HBM traffic = 268MB fp32 query stream
// => memory bound (~11.5us @ 23.3TB/s). Strategy: LN support once into an
// L2-resident bf16 workspace (32x208x1024 bf16 = 13.6MB, padded to 208 rows
// per batch so the 13th 16-wide WMMA B-tile is valid), then one fused
// wave32 kernel per 16-query-row tile: LN stats pass + WMMA k-loop that
// normalizes+converts the (cache-hot) query tile to bf16 on the fly, and an
// LDS-staged max/min/argmax epilogue. Query streams from HBM exactly once.
// ---------------------------------------------------------------------------

typedef __attribute__((ext_vector_type(16))) __bf16 v16bf;
typedef __attribute__((ext_vector_type(8)))  float  v8f;

#define HDIM  1024
#define NCLS  20
#define KSH   10
#define NK    200     // NCLS*KSH
#define NKP   208     // padded to 13 * 16
#define NTIL  13      // NKP / 16
#define QTOT  2048
#define NBAT  32
#define LNEPS 1e-5f

__device__ __forceinline__ unsigned short f2bf(float f) {
  union { float f; unsigned u; } v; v.f = f;
  unsigned u = v.u;
  u += 0x7fffu + ((u >> 16) & 1u);   // round-to-nearest-even
  return (unsigned short)(u >> 16);
}

// ---------------------------------------------------------------------------
// Kernel A: LayerNorm support rows -> bf16 workspace, padded rows zeroed.
// One wave per output row (4 waves / 128-thread block).
// ---------------------------------------------------------------------------
__global__ void __launch_bounds__(128)
ln_support_kernel(const float* __restrict__ sup,
                  const float* __restrict__ gamma,
                  const float* __restrict__ beta,
                  unsigned short* __restrict__ sws) {
  int wave = blockIdx.x * (blockDim.x >> 5) + (threadIdx.x >> 5);
  int lane = threadIdx.x & 31;
  if (wave >= NBAT * NKP) return;
  int b = wave / NKP;
  int r = wave - b * NKP;
  unsigned short* dst = sws + (size_t)wave * HDIM;

  if (r >= NK) {                       // zero-pad fake rows 200..207
    ushort4 z = make_ushort4(0, 0, 0, 0);
    #pragma unroll
    for (int i = 0; i < 8; ++i)
      *(ushort4*)(dst + i * 128 + lane * 4) = z;
    return;
  }

  const float* src = sup + ((size_t)b * NK + r) * HDIM;
  float4 x[8];
  float s = 0.f, ss = 0.f;
  #pragma unroll
  for (int i = 0; i < 8; ++i) {
    x[i] = *(const float4*)(src + i * 128 + lane * 4);
    s += x[i].x + x[i].y + x[i].z + x[i].w;
    ss = fmaf(x[i].x, x[i].x, fmaf(x[i].y, x[i].y,
         fmaf(x[i].z, x[i].z, fmaf(x[i].w, x[i].w, ss))));
  }
  #pragma unroll
  for (int m = 16; m >= 1; m >>= 1) {
    s  += __shfl_xor(s,  m, 32);
    ss += __shfl_xor(ss, m, 32);
  }
  float mean = s  * (1.f / HDIM);
  float var  = ss * (1.f / HDIM) - mean * mean;   // biased variance
  float rstd = rsqrtf(var + LNEPS);

  #pragma unroll
  for (int i = 0; i < 8; ++i) {
    int h = i * 128 + lane * 4;
    float4 g  = *(const float4*)(gamma + h);
    float4 bb = *(const float4*)(beta  + h);
    ushort4 o;
    o.x = f2bf(fmaf((x[i].x - mean) * rstd, g.x, bb.x));
    o.y = f2bf(fmaf((x[i].y - mean) * rstd, g.y, bb.y));
    o.z = f2bf(fmaf((x[i].z - mean) * rstd, g.z, bb.z));
    o.w = f2bf(fmaf((x[i].w - mean) * rstd, g.w, bb.w));
    *(ushort4*)(dst + h) = o;
  }
}

// ---------------------------------------------------------------------------
// Kernel B: fused query-LN + WMMA GEMM + max/min/argmax epilogue.
// One wave32 per 16 query rows. 4096 blocks total.
// ---------------------------------------------------------------------------
__global__ void __launch_bounds__(32)
fused_qln_gemm_kernel(const float* __restrict__ qry,
                      const float* __restrict__ gamma,
                      const float* __restrict__ beta,
                      const unsigned short* __restrict__ sws,
                      float* __restrict__ logits_out,
                      float* __restrict__ pred_out) {
  __shared__ float lds_g[HDIM];
  __shared__ float lds_b[HDIM];
  __shared__ float lds_d[16 * 212];          // D staging, padded stride

  const int t     = blockIdx.x;              // 16-row query tile id
  const int lane  = threadIdx.x;
  const int batch = t >> 7;                  // t / (QTOT/16)
  const int m     = lane & 15;
  const int half  = lane >> 4;

  // cache gamma/beta in LDS (reused 32x by the k-loop)
  #pragma unroll
  for (int i = 0; i < 8; ++i) {
    int h = i * 128 + lane * 4;
    *(float4*)(lds_g + h) = *(const float4*)(gamma + h);
    *(float4*)(lds_b + h) = *(const float4*)(beta  + h);
  }

  // ---- LN stats: lane pair (l, l^16) covers row m, 512 floats each -------
  const float* qrow = qry + ((size_t)t * 16 + m) * HDIM;
  float mean, rstd;
  {
    const float* p = qrow + half * 512;
    float s = 0.f, ss = 0.f;
    for (int i = 0; i < 128; ++i) {
      float4 x = *(const float4*)(p + i * 4);
      s += x.x + x.y + x.z + x.w;
      ss = fmaf(x.x, x.x, fmaf(x.y, x.y, fmaf(x.z, x.z, fmaf(x.w, x.w, ss))));
    }
    s  += __shfl_xor(s,  16, 32);
    ss += __shfl_xor(ss, 16, 32);
    mean = s  * (1.f / HDIM);
    float var = ss * (1.f / HDIM) - mean * mean;
    rstd = rsqrtf(var + LNEPS);
  }

  // ---- WMMA k-loop: 32 chunks of K=32, 13 B-tiles ------------------------
  v8f acc[NTIL];
  #pragma unroll
  for (int nt = 0; nt < NTIL; ++nt) acc[nt] = (v8f)(0.f);

  // A layout (16x32 bf16): lane m holds K = {0..7} + 8*half and {16..23} + 8*half
  // B layout (32x16 bf16): lane n holds K = {0..7} + 8*half and {16..23} + 8*half
  const int ka = half * 8;
  const unsigned short* sbase =
      sws + ((size_t)batch * NKP + m) * HDIM + half * 8;

  for (int kc = 0; kc < 32; ++kc) {
    const int k0 = kc * 32 + ka;
    float4 a0 = *(const float4*)(qrow + k0);
    float4 a1 = *(const float4*)(qrow + k0 + 4);
    float4 a2 = *(const float4*)(qrow + k0 + 16);
    float4 a3 = *(const float4*)(qrow + k0 + 20);
    float4 g0 = *(const float4*)(lds_g + k0);
    float4 g1 = *(const float4*)(lds_g + k0 + 4);
    float4 g2 = *(const float4*)(lds_g + k0 + 16);
    float4 g3 = *(const float4*)(lds_g + k0 + 20);
    float4 b0 = *(const float4*)(lds_b + k0);
    float4 b1 = *(const float4*)(lds_b + k0 + 4);
    float4 b2 = *(const float4*)(lds_b + k0 + 16);
    float4 b3 = *(const float4*)(lds_b + k0 + 20);

    union { v16bf v; unsigned short h[16]; } A;
    A.h[0]  = f2bf(fmaf((a0.x - mean) * rstd, g0.x, b0.x));
    A.h[1]  = f2bf(fmaf((a0.y - mean) * rstd, g0.y, b0.y));
    A.h[2]  = f2bf(fmaf((a0.z - mean) * rstd, g0.z, b0.z));
    A.h[3]  = f2bf(fmaf((a0.w - mean) * rstd, g0.w, b0.w));
    A.h[4]  = f2bf(fmaf((a1.x - mean) * rstd, g1.x, b1.x));
    A.h[5]  = f2bf(fmaf((a1.y - mean) * rstd, g1.y, b1.y));
    A.h[6]  = f2bf(fmaf((a1.z - mean) * rstd, g1.z, b1.z));
    A.h[7]  = f2bf(fmaf((a1.w - mean) * rstd, g1.w, b1.w));
    A.h[8]  = f2bf(fmaf((a2.x - mean) * rstd, g2.x, b2.x));
    A.h[9]  = f2bf(fmaf((a2.y - mean) * rstd, g2.y, b2.y));
    A.h[10] = f2bf(fmaf((a2.z - mean) * rstd, g2.z, b2.z));
    A.h[11] = f2bf(fmaf((a2.w - mean) * rstd, g2.w, b2.w));
    A.h[12] = f2bf(fmaf((a3.x - mean) * rstd, g3.x, b3.x));
    A.h[13] = f2bf(fmaf((a3.y - mean) * rstd, g3.y, b3.y));
    A.h[14] = f2bf(fmaf((a3.z - mean) * rstd, g3.z, b3.z));
    A.h[15] = f2bf(fmaf((a3.w - mean) * rstd, g3.w, b3.w));

    const unsigned short* sp = sbase + kc * 32;
    #pragma unroll
    for (int nt = 0; nt < NTIL; ++nt) {
      union { v16bf v; uint4 q[2]; } Bf;
      Bf.q[0] = *(const uint4*)(sp + (size_t)nt * 16 * HDIM);
      Bf.q[1] = *(const uint4*)(sp + (size_t)nt * 16 * HDIM + 16);
      acc[nt] = __builtin_amdgcn_wmma_f32_16x16x32_bf16(
          false, A.v, false, Bf.v, (short)0, acc[nt], false, false);
    }
  }

  // ---- epilogue: stage D (M=r+8*half, N=m) then reduce ------------------
  #pragma unroll
  for (int nt = 0; nt < NTIL; ++nt) {
    #pragma unroll
    for (int r = 0; r < 8; ++r)
      lds_d[(r + 8 * half) * 212 + nt * 16 + m] = acc[nt][r];
  }
  __syncthreads();

  if (lane < 16) {
    const float* drow = lds_d + lane * 212;
    float lg[NCLS];
    float minv  = INFINITY;
    float bestv = -INFINITY;
    int   best  = 0;
    #pragma unroll
    for (int c = 0; c < NCLS; ++c) {
      float mx = drow[c * KSH];
      #pragma unroll
      for (int k = 1; k < KSH; ++k) mx = fmaxf(mx, drow[c * KSH + k]);
      lg[c] = mx;
      minv  = fminf(minv, mx);
      if (mx > bestv) { bestv = mx; best = c; }   // first-occurrence argmax
    }
    size_t row = (size_t)t * 16 + lane;
    float* o = logits_out + row * (NCLS + 1);
    #pragma unroll
    for (int c = 0; c < NCLS; ++c) o[c] = lg[c];
    o[NCLS] = minv - 1.f;                 // always < all logits -> never argmax
    pred_out[row] = (float)best;
  }
}

// ---------------------------------------------------------------------------
extern "C" void kernel_launch(void* const* d_in, const int* in_sizes, int n_in,
                              void* d_out, int out_size, void* d_ws, size_t ws_size,
                              hipStream_t stream) {
  const float* support = (const float*)d_in[0];
  const float* query   = (const float*)d_in[1];
  const float* gamma   = (const float*)d_in[2];
  const float* beta    = (const float*)d_in[3];

  float* logits = (float*)d_out;                              // (B, Q, N+1)
  float* pred   = logits + (size_t)NBAT * QTOT * (NCLS + 1);  // (B*Q,) as float

  unsigned short* sws = (unsigned short*)d_ws;  // 32*208*1024 bf16 = 13.6 MB

  // support LN -> bf16 workspace (pads rows 200..207 with zeros)
  ln_support_kernel<<<(NBAT * NKP) / 4, 128, 0, stream>>>(support, gamma, beta, sws);

  // fused query-LN + GEMM + epilogue: one wave per 16 query rows
  fused_qln_gemm_kernel<<<NBAT * QTOT / 16, 32, 0, stream>>>(
      query, gamma, beta, sws, logits, pred);
}
